// MultiHeadAttentionViz_38508676776566
// MI455X (gfx1250) — matrix operand order repaired and checked by reading, not compiled, not run on hardware
//
#include <hip/hip_runtime.h>
#include <hip/hip_bf16.h>

typedef __attribute__((ext_vector_type(16))) _Float16 v16h;
typedef __attribute__((ext_vector_type(8)))  float    v8f;

constexpr int B_ = 2, T_ = 2048, D_ = 1024, H_ = 16, DK_ = 64;
constexpr int M_ = B_ * T_;                  // 4096

// GEMM tiling
constexpr int BM = 128, BN = 128, BK = 64;
constexpr int LDT = BK + 8;                  // 72 halves = 144B stride (16B aligned, bank-spread)

// ---------------------------------------------------------------------------
// gfx1250 async global->LDS copy (ASYNCcnt), 16B per lane.
// LDS address operand = wave-relative LDS offset = low 32 bits of generic ptr.
// ---------------------------------------------------------------------------
__device__ __forceinline__ void cp_async16(void* lds_dst, const void* gsrc) {
  unsigned int loff = (unsigned int)(unsigned long long)lds_dst;
  asm volatile("global_load_async_to_lds_b128 %0, %1, off"
               :: "v"(loff), "v"(gsrc)
               : "memory");
}
__device__ __forceinline__ void wait_async0() {
#if __has_builtin(__builtin_amdgcn_s_wait_asynccnt)
  __builtin_amdgcn_s_wait_asynccnt(0);
#else
  asm volatile("s_wait_asynccnt 0x0" ::: "memory");
#endif
}

__device__ __forceinline__ v8f wmma_f16(v16h a, v16h b, v8f c) {
  return __builtin_amdgcn_wmma_f32_16x16x32_f16(false, a, false, b, (short)0, c,
                                                false, false);
}

// ---------------------------------------------------------------------------
// WMMA fragment loaders from LDS (this code's contiguous k order, same for A and B, wave32)
// A 16x32 f16: lane L -> row base+(L&15); halves: lanes<16: K{0..15},
//              lanes>=16: K{16..31}
// B 32x16 f16: lane L -> col base+(L&15); halves: K{0..15} / K{16..31}
// ---------------------------------------------------------------------------
__device__ __forceinline__ v16h ld_frag_a(const _Float16* base, int lda, int row0,
                                          int kk, int lane) {
  const _Float16* r = base + (row0 + (lane & 15)) * lda + kk;
  const int g = (lane >> 4) * 16;
  union { v16h v; unsigned long long u[4]; } f;
  const unsigned long long* p0 = (const unsigned long long*)(r + g);
  const unsigned long long* p1 = (const unsigned long long*)(r + 8 + g);
  f.u[0] = p0[0]; f.u[1] = p0[1];
  f.u[2] = p1[0]; f.u[3] = p1[1];
  return f.v;
}

__device__ __forceinline__ v16h ld_frag_b(const _Float16* base, int lda, int col0,
                                          int kk, int lane) {
  const _Float16* r = base + (col0 + (lane & 15)) * lda + kk + (lane >> 4) * 16;
  union { v16h v; unsigned long long u[4]; } f;
  const unsigned long long* p = (const unsigned long long*)r;
  f.u[0] = p[0]; f.u[1] = p[1]; f.u[2] = p[2]; f.u[3] = p[3];
  return f.v;
}

// ---------------------------------------------------------------------------
// f32 -> f16 conversion (bandwidth-bound, negligible vs compute)
// ---------------------------------------------------------------------------
__global__ __launch_bounds__(256) void cvt_f16(const float* __restrict__ src,
                                               _Float16* __restrict__ dst, int n4) {
  int i = blockIdx.x * 256 + threadIdx.x;
  if (i < n4) {
    float4 v = *(const float4*)(src + (size_t)i * 4);
    union { _Float16 h[4]; unsigned long long u; } t;
    t.h[0] = (_Float16)v.x; t.h[1] = (_Float16)v.y;
    t.h[2] = (_Float16)v.z; t.h[3] = (_Float16)v.w;
    *(unsigned long long*)(dst + (size_t)i * 4) = t.u;
  }
}

// ---------------------------------------------------------------------------
// Tiled GEMM: out[m,n] = sum_k A[m,k] * W[n,k] + bias[n]   (torch Linear, NT)
// A and W are f16. Async double-buffered LDS staging.
// AMODE 0: A f16 [M_,D_] row-major   AMODE 1: A f16 ctx [B,H,T,DK] layout
// OMODE 0: f16 -> [B,H,T,DK]   OMODE 1: f16 -> [B,H,DK,T]   OMODE 2: f32 + bias
// ---------------------------------------------------------------------------
template <int AMODE, int OMODE>
__global__ __launch_bounds__(256) void gemm_wmma(const _Float16* __restrict__ Af,
                                                 const _Float16* __restrict__ Wf,
                                                 const float* __restrict__ bias,
                                                 void* __restrict__ Outp) {
  __shared__ __align__(16) _Float16 shA[2][BM * LDT];
  __shared__ __align__(16) _Float16 shB[2][BN * LDT];

  const int tid  = threadIdx.x;
  const int lane = tid & 31;
  const int w    = tid >> 5;       // 8 waves
  const int wm   = w & 3;          // 4 waves along M -> 32 rows each
  const int wn   = w >> 2;         // 2 waves along N -> 64 cols each
  const int m0   = blockIdx.y * BM;
  const int n0   = blockIdx.x * BN;

  auto stage = [&](int buf, int k0) {
#pragma unroll
    for (int i = 0; i < 4; ++i) {
      int slot = tid + i * 256;              // 1024 chunks of 8 halves
      int row = slot >> 3, c8 = slot & 7;
      const _Float16* asrc;
      if constexpr (AMODE == 0) {
        asrc = Af + (size_t)(m0 + row) * D_ + k0 + c8 * 8;
      } else {
        int m = m0 + row, b = m >> 11, t = m & (T_ - 1), hh = k0 >> 6;
        asrc = Af + ((size_t)(b * H_ + hh) * T_ + t) * DK_ + c8 * 8;
      }
      cp_async16(&shA[buf][row * LDT + c8 * 8], asrc);
      cp_async16(&shB[buf][row * LDT + c8 * 8],
                 Wf + (size_t)(n0 + row) * D_ + k0 + c8 * 8);
    }
  };

  v8f acc[2][4] = {};

  stage(0, 0);
  wait_async0();
  __syncthreads();

  int cur = 0;
  for (int k0 = 0; k0 < D_; k0 += BK) {
    if (k0 + BK < D_) stage(cur ^ 1, k0 + BK);  // overlap with math below
#pragma unroll
    for (int kk = 0; kk < BK; kk += 32) {
      // batch ALL fragment loads, then run the 8-WMMA chain back-to-back
      v16h af0 = ld_frag_a(shA[cur], LDT, wm * 32, kk, lane);
      v16h af1 = ld_frag_a(shA[cur], LDT, wm * 32 + 16, kk, lane);
      v16h bf[4];
#pragma unroll
      for (int ni = 0; ni < 4; ++ni)
        bf[ni] = ld_frag_b(shB[cur], LDT, wn * 64 + ni * 16, kk, lane);
#pragma unroll
      for (int ni = 0; ni < 4; ++ni) {
        acc[0][ni] = wmma_f16(af0, bf[ni], acc[0][ni]);
        acc[1][ni] = wmma_f16(af1, bf[ni], acc[1][ni]);
      }
    }
    wait_async0();     // next tile landed in LDS
    __syncthreads();   // everyone done with current tile
    cur ^= 1;
  }

  // ---- epilogue: C layout -> VGPR e: M=e+8*(lane>=16), N=lane&15 ----
  if constexpr (OMODE == 1) {
    // lane holds 8 consecutive t for fixed dk -> vectorized 16B store
#pragma unroll
    for (int mi = 0; mi < 2; ++mi) {
#pragma unroll
      for (int ni = 0; ni < 4; ++ni) {
        int n = n0 + wn * 64 + ni * 16 + (lane & 15);
        float bv = bias[n];
        union { _Float16 h[8]; unsigned long long u[2]; } t2;
#pragma unroll
        for (int e = 0; e < 8; ++e) t2.h[e] = (_Float16)(acc[mi][ni][e] + bv);
        int mb = m0 + wm * 32 + mi * 16 + ((lane >> 4) << 3);
        int b = mb >> 11, tt = mb & (T_ - 1);
        int hh = n >> 6, dk = n & 63;
        unsigned long long* dst = (unsigned long long*)(
            (_Float16*)Outp + ((size_t)(b * H_ + hh) * DK_ + dk) * T_ + tt);
        dst[0] = t2.u[0]; dst[1] = t2.u[1];
      }
    }
  } else {
#pragma unroll
    for (int mi = 0; mi < 2; ++mi) {
#pragma unroll
      for (int ni = 0; ni < 4; ++ni) {
#pragma unroll
        for (int e = 0; e < 8; ++e) {
          int m = m0 + wm * 32 + mi * 16 + e + ((lane >> 4) << 3);
          int n = n0 + wn * 64 + ni * 16 + (lane & 15);
          float v = acc[mi][ni][e] + bias[n];
          if constexpr (OMODE == 2) {
            ((float*)Outp)[(size_t)m * D_ + n] = v;
          } else {
            int b = m >> 11, t = m & (T_ - 1);
            int hh = n >> 6, dk = n & 63;
            ((_Float16*)Outp)[((size_t)(b * H_ + hh) * T_ + t) * DK_ + dk] =
                (_Float16)v;
          }
        }
      }
    }
  }
}

// ---------------------------------------------------------------------------
// Flash attention: grid (T/64, B*H), 128 threads (4 waves x 16 query rows).
// Async double-buffered K/V/mask tiles; online softmax; WMMA for S and P@V.
// ---------------------------------------------------------------------------
__global__ __launch_bounds__(128) void attn_wmma(const _Float16* __restrict__ qh,
                                                 const _Float16* __restrict__ kh,
                                                 const _Float16* __restrict__ vth,
                                                 const int* __restrict__ maskp,
                                                 _Float16* __restrict__ ctx) {
  constexpr int LQ = DK_ + 8;  // 72
  __shared__ __align__(16) _Float16 Qt[64 * LQ];
  __shared__ __align__(16) _Float16 Kt[2][64 * LQ];
  __shared__ __align__(16) _Float16 Vt[2][64 * LQ];   // V^T tile: [d][s]
  __shared__ __align__(16) _Float16 Pt[4 * 16 * LQ];  // per-wave P staging
  __shared__ __align__(16) int      Mt[2][64];        // key-padding mask tile

  const int tid  = threadIdx.x;
  const int lane = tid & 31;
  const int w    = tid >> 5;
  const int bh   = blockIdx.y;
  const int b    = bh >> 4;
  const int t0   = blockIdx.x * 64;

  const _Float16* qb = qh + (size_t)bh * T_ * DK_;
  const _Float16* kb = kh + (size_t)bh * T_ * DK_;
  const _Float16* vb = vth + (size_t)bh * DK_ * T_;
  const int*      mb = maskp + b * T_;

  auto stage_kv = [&](int buf, int s0) {
#pragma unroll
    for (int i = 0; i < 4; ++i) {
      int slot = tid + i * 128;  // 512 chunks of 8 halves per tile
      int row = slot >> 3, c8 = slot & 7;
      cp_async16(&Kt[buf][row * LQ + c8 * 8],
                 kb + (size_t)(s0 + row) * DK_ + c8 * 8);
      cp_async16(&Vt[buf][row * LQ + c8 * 8],
                 vb + (size_t)row * T_ + s0 + c8 * 8);
    }
    if (tid < 16)  // 64 ints = 16 x 16B chunks
      cp_async16(&Mt[buf][tid * 4], mb + s0 + tid * 4);
  };

  // stage Q (once) + first K/V/mask tile
#pragma unroll
  for (int i = 0; i < 4; ++i) {
    int slot = tid + i * 128;
    int row = slot >> 3, c8 = slot & 7;
    cp_async16(&Qt[row * LQ + c8 * 8], qb + (size_t)(t0 + row) * DK_ + c8 * 8);
  }
  stage_kv(0, 0);
  wait_async0();
  __syncthreads();

  v16h qf[2];
  qf[0] = ld_frag_a(Qt, LQ, w * 16, 0, lane);
  qf[1] = ld_frag_a(Qt, LQ, w * 16, 32, lane);

  float rowM[8], rowL[8];
  v8f o[4] = {};
#pragma unroll
  for (int e = 0; e < 8; ++e) { rowM[e] = -3.0e38f; rowL[e] = 0.f; }
  const float scale = 0.125f;  // 1/sqrt(64)

  int cur = 0;
  for (int s0 = 0; s0 < T_; s0 += 64) {
    if (s0 + 64 < T_) stage_kv(cur ^ 1, s0 + 64);  // overlap with math

    // mask flags from LDS (no global waits in the hot loop)
    bool keep[4];
#pragma unroll
    for (int ni = 0; ni < 4; ++ni)
      keep[ni] = Mt[cur][ni * 16 + (lane & 15)] != 0;

    // S = Q * K^T : batch all 8 B-fragment loads, then run the WMMA chains
    v16h bfa[4][2];
#pragma unroll
    for (int ni = 0; ni < 4; ++ni)
#pragma unroll
      for (int kk = 0; kk < 2; ++kk)
        bfa[ni][kk] = ld_frag_b(Kt[cur], LQ, ni * 16, kk * 32, lane);

    float sv[4][8];
#pragma unroll
    for (int ni = 0; ni < 4; ++ni) {
      v8f s_acc = {};
      s_acc = wmma_f16(qf[0], bfa[ni][0], s_acc);
      s_acc = wmma_f16(qf[1], bfa[ni][1], s_acc);
#pragma unroll
      for (int e = 0; e < 8; ++e)
        sv[ni][e] = keep[ni] ? s_acc[e] * scale : -1.0e30f;
    }

    // online softmax (row lives in 16 lanes; xor<=8 stays within the half)
#pragma unroll
    for (int e = 0; e < 8; ++e) {
      float mx = fmaxf(fmaxf(sv[0][e], sv[1][e]), fmaxf(sv[2][e], sv[3][e]));
#pragma unroll
      for (int off = 8; off >= 1; off >>= 1) mx = fmaxf(mx, __shfl_xor(mx, off, 32));
      float nM = fmaxf(rowM[e], mx);
      float cf = __expf(rowM[e] - nM);
      rowM[e] = nM;
      float rs = 0.f;
#pragma unroll
      for (int ni = 0; ni < 4; ++ni) {
        float p = __expf(sv[ni][e] - nM);
        sv[ni][e] = p;
        rs += p;
      }
#pragma unroll
      for (int off = 8; off >= 1; off >>= 1) rs += __shfl_xor(rs, off, 32);
      rowL[e] = rowL[e] * cf + rs;
#pragma unroll
      for (int ni = 0; ni < 4; ++ni) o[ni][e] *= cf;
    }

    // P -> per-wave LDS region (same-wave DS ops are in-order; no barrier)
    _Float16* Pw = &Pt[w * 16 * LQ];
#pragma unroll
    for (int ni = 0; ni < 4; ++ni)
#pragma unroll
      for (int e = 0; e < 8; ++e) {
        int r = e + ((lane >> 4) << 3);
        Pw[r * LQ + ni * 16 + (lane & 15)] = (_Float16)sv[ni][e];
      }

    // O += P @ V : batch fragment loads per k-step, then the 4-WMMA chain
#pragma unroll
    for (int kk = 0; kk < 2; ++kk) {
      v16h af = ld_frag_a(Pw, LQ, 0, kk * 32, lane);
      v16h bf[4];
#pragma unroll
      for (int ni = 0; ni < 4; ++ni)
        bf[ni] = ld_frag_b(Vt[cur], LQ, ni * 16, kk * 32, lane);
#pragma unroll
      for (int ni = 0; ni < 4; ++ni) o[ni] = wmma_f16(af, bf[ni], o[ni]);
    }

    wait_async0();     // next K/V/mask tile landed
    __syncthreads();   // all waves done with current tile
    cur ^= 1;
  }

  // normalize, store ctx f16 in [B,H,T,DK]
#pragma unroll
  for (int e = 0; e < 8; ++e) {
    float inv = rowL[e] > 0.f ? 1.0f / rowL[e] : 0.f;
    int t = t0 + w * 16 + e + ((lane >> 4) << 3);
#pragma unroll
    for (int ni = 0; ni < 4; ++ni) {
      int d = ni * 16 + (lane & 15);
      ctx[((size_t)bh * T_ + t) * DK_ + d] = (_Float16)(o[ni][e] * inv);
    }
  }
}

// ---------------------------------------------------------------------------
extern "C" void kernel_launch(void* const* d_in, const int* in_sizes, int n_in,
                              void* d_out, int out_size, void* d_ws, size_t ws_size,
                              hipStream_t stream) {
  const float* query = (const float*)d_in[0];
  const float* key   = (const float*)d_in[1];
  const float* value = (const float*)d_in[2];
  const int*   mask  = (const int*)d_in[3];  // True = keep
  const float* Wq = (const float*)d_in[4];
  const float* bq = (const float*)d_in[5];
  const float* Wk = (const float*)d_in[6];
  const float* bk = (const float*)d_in[7];
  const float* Wv = (const float*)d_in[8];
  const float* bv = (const float*)d_in[9];
  const float* Wo = (const float*)d_in[10];
  const float* bo = (const float*)d_in[11];

  const size_t act = (size_t)M_ * D_;        // 4M elems
  const size_t wel = (size_t)D_ * D_;        // 1M elems
  _Float16* qx  = (_Float16*)d_ws;           // f16 activations
  _Float16* kx  = qx + act;
  _Float16* vx  = kx + act;
  _Float16* wqh = vx + act;                  // f16 weights
  _Float16* wkh = wqh + wel;
  _Float16* wvh = wkh + wel;
  _Float16* woh = wvh + wel;
  _Float16* qh  = woh + wel;                 // head-layout intermediates
  _Float16* kh  = qh + act;
  _Float16* vth = kh + act;
  _Float16* ctx = vth + act;                 // total: 64 MB

  // f32 -> f16 passes (memory-bound)
  cvt_f16<<<dim3(act / 1024), 256, 0, stream>>>(query, qx, (int)(act / 4));
  cvt_f16<<<dim3(act / 1024), 256, 0, stream>>>(key, kx, (int)(act / 4));
  cvt_f16<<<dim3(act / 1024), 256, 0, stream>>>(value, vx, (int)(act / 4));
  cvt_f16<<<dim3(wel / 1024), 256, 0, stream>>>(Wq, wqh, (int)(wel / 4));
  cvt_f16<<<dim3(wel / 1024), 256, 0, stream>>>(Wk, wkh, (int)(wel / 4));
  cvt_f16<<<dim3(wel / 1024), 256, 0, stream>>>(Wv, wvh, (int)(wel / 4));
  cvt_f16<<<dim3(wel / 1024), 256, 0, stream>>>(Wo, woh, (int)(wel / 4));

  dim3 gg(D_ / BN, M_ / BM);  // (8, 32), 256 threads
  gemm_wmma<0, 0><<<gg, 256, 0, stream>>>(qx, wqh, bq, qh);
  gemm_wmma<0, 0><<<gg, 256, 0, stream>>>(kx, wkh, bk, kh);
  gemm_wmma<0, 1><<<gg, 256, 0, stream>>>(vx, wvh, bv, vth);

  dim3 ga(T_ / 64, B_ * H_);  // (32, 32), 128 threads
  attn_wmma<<<ga, 128, 0, stream>>>(qh, kh, vth, mask, ctx);

  gemm_wmma<1, 2><<<gg, 256, 0, stream>>>(ctx, woh, bo, d_out);
}
